// Pix2StructVisionAttention_42107859370155
// MI455X (gfx1250) — matrix operand order repaired and checked by reading, not compiled, not run on hardware
//
#include <hip/hip_runtime.h>
#include <hip/hip_bf16.h>

typedef __bf16 bf16;
typedef __attribute__((ext_vector_type(16))) __bf16 v16bf;
typedef __attribute__((ext_vector_type(8)))  __bf16 v8bf;
typedef __attribute__((ext_vector_type(4)))  __bf16 v4bf;
typedef __attribute__((ext_vector_type(8)))  float  v8f;
typedef __attribute__((ext_vector_type(4)))  int    v4i;

#define NEGMAX (-3.402823466e38f)

// ---------------------------------------------------------------------------
// CDNA5 async global->LDS staging (ASYNCcnt-tracked), with safe fallback.
// Probe-confirmed: builtin takes (v4i AS1*, v4i AS3*, imm offset, imm cpol).
// ---------------------------------------------------------------------------
#if defined(__AMDGCN__) && __has_builtin(__builtin_amdgcn_global_load_async_to_lds_b128)
#define USE_ASYNC_LDS 1
#else
#define USE_ASYNC_LDS 0
#endif

typedef __attribute__((address_space(1))) v4i* gv4i_p;
typedef __attribute__((address_space(3))) v4i* lv4i_p;

static __device__ __forceinline__ void cp16(bf16* l, const bf16* g) {
#if USE_ASYNC_LDS
  __builtin_amdgcn_global_load_async_to_lds_b128((gv4i_p)(void*)g, (lv4i_p)(void*)l,
                                                 0, 0);
#else
  *(v8bf*)l = *(const v8bf*)g;
#endif
}
static __device__ __forceinline__ void wait_async0() {
#if USE_ASYNC_LDS
#if __has_builtin(__builtin_amdgcn_s_wait_asynccnt)
  __builtin_amdgcn_s_wait_asynccnt(0);
#else
  asm volatile("s_wait_asynccnt 0x0" ::: "memory");
#endif
#endif
}

// ---------------------------------------------------------------------------
// WMMA helpers (CDNA5 16x16x32 bf16, f32 accum).
// A-operand (16x32): lane M = lane&15, half = lane>>4;
//   elems = A[M][k0+8h .. +7] ++ A[M][k0+16+8h .. +7]
// B-operand (32x16): lane N = lane&15; elems = B[k0+8h .. +7][N] ++
//   B[k0+16+8h .. +7][N] (tile stored [N][K])
// C/D (16x16 f32): elem v -> row (v + 8*half), col (lane&15)
// ---------------------------------------------------------------------------
static __device__ __forceinline__ v16bf join8(v8bf lo, v8bf hi) {
  return __builtin_shufflevector(lo, hi, 0,1,2,3,4,5,6,7,8,9,10,11,12,13,14,15);
}
static __device__ __forceinline__ v8f wmma_bf16(v16bf a, v16bf b, v8f c) {
  return __builtin_amdgcn_wmma_f32_16x16x32_bf16(false, a, false, b, (short)0, c,
                                                 false, false);
}
static __device__ __forceinline__ v16bf frag_a(const bf16* t, int stride, int row,
                                               int k0, int half) {
  const bf16* p = t + row * stride + k0 + half * 8;
  return join8(*(const v8bf*)p, *(const v8bf*)(p + 16));
}
static __device__ __forceinline__ v16bf frag_b(const bf16* t, int stride, int row,
                                               int k0, int half) {
  const bf16* p = t + row * stride + k0 + half * 8;
  return join8(*(const v8bf*)p, *(const v8bf*)(p + 16));
}

// ---------------------------------------------------------------------------
// f32 -> bf16 conversion (4 elements / thread)
// ---------------------------------------------------------------------------
__global__ __launch_bounds__(256) void cvt_f32_bf16(const float* __restrict__ s,
                                                    bf16* __restrict__ d) {
  int i = (blockIdx.x * 256 + threadIdx.x) * 4;
  float4 f = *(const float4*)(s + i);
  v4bf o;
  o[0] = (bf16)f.x; o[1] = (bf16)f.y; o[2] = (bf16)f.z; o[3] = (bf16)f.w;
  *(v4bf*)(d + i) = o;
}

// ---------------------------------------------------------------------------
// Y = X * W^T   (X: [4096 x 1024] bf16, W: [1024 x 1024] bf16 row-major)
// mode 0: bf16 [2][16][2048][64]  (Q, K)
// mode 1: bf16 [2][16][64][2048]  (V transposed; packed b128 stores)
// mode 2: f32  [4096][1024]       (final output)
// grid = 128 blocks (8 N-tiles x 16 M-tiles of 256), 256 threads (8 waves).
// Wave tile 32(M)x128(N): 2 A-frags amortize each B-frag -> 1.25 ds/WMMA.
// ---------------------------------------------------------------------------
__global__ __launch_bounds__(256) void gemm_x_wT(const bf16* __restrict__ A,
                                                 const bf16* __restrict__ W,
                                                 void* __restrict__ outp,
                                                 int mode) {
  __shared__ __align__(16) bf16 As[2][256][40];
  __shared__ __align__(16) bf16 Bs[2][128][40];
  const int t    = threadIdx.x;
  const int gx   = blockIdx.x & 7;
  const int gy   = blockIdx.x >> 3;
  const int m0   = gy * 256, n0 = gx * 128;
  const int wid  = t >> 5, lane = t & 31, ln = lane & 15, half = lane >> 4;

  auto load_tile = [&](int buf, int kb) {
    // A: 256 rows x 32 k — one full row per thread
    const bf16* ga = A + (size_t)(m0 + t) * 1024 + kb * 32;
    cp16(&As[buf][t][0],  ga);
    cp16(&As[buf][t][8],  ga + 8);
    cp16(&As[buf][t][16], ga + 16);
    cp16(&As[buf][t][24], ga + 24);
    // B: 128 rows x 32 k — half row per thread
    int br = t >> 1, bc = (t & 1) * 16;
    const bf16* gw = W + (size_t)(n0 + br) * 1024 + kb * 32 + bc;
    cp16(&Bs[buf][br][bc],     gw);
    cp16(&Bs[buf][br][bc + 8], gw + 8);
  };

  v8f acc[2][8] = {};
  load_tile(0, 0);
  for (int kb = 0; kb < 32; ++kb) {
    wait_async0();
    __syncthreads();
    if (kb + 1 < 32) load_tile((kb + 1) & 1, kb + 1);
    const bf16* as = &As[kb & 1][0][0];
    const bf16* bs = &Bs[kb & 1][0][0];
    v16bf a0 = frag_a(as, 40, wid * 32 + ln, 0, half);
    v16bf a1 = frag_a(as, 40, wid * 32 + 16 + ln, 0, half);
#pragma unroll
    for (int nt = 0; nt < 8; ++nt) {
      v16bf b = frag_b(bs, 40, nt * 16 + ln, 0, half);
      acc[0][nt] = wmma_bf16(a0, b, acc[0][nt]);
      acc[1][nt] = wmma_bf16(a1, b, acc[1][nt]);
    }
  }

  if (mode == 0) {                       // bf16 [b][h][s][d]
    bf16* o = (bf16*)outp;
#pragma unroll
    for (int mi = 0; mi < 2; ++mi)
#pragma unroll
      for (int nt = 0; nt < 8; ++nt) {
        int n = n0 + nt * 16 + ln, hh = n >> 6, dd = n & 63;
#pragma unroll
        for (int v = 0; v < 8; ++v) {
          int m = m0 + wid * 32 + mi * 16 + v + 8 * half;
          int bb = m >> 11, s = m & 2047;
          o[(((size_t)bb * 16 + hh) * 2048 + s) * 64 + dd] = (bf16)acc[mi][nt][v];
        }
      }
  } else if (mode == 1) {                // bf16 [b][h][d][s] — packed store
    bf16* o = (bf16*)outp;
#pragma unroll
    for (int mi = 0; mi < 2; ++mi) {
      int m = m0 + wid * 32 + mi * 16 + 8 * half;  // s for v=0; v contiguous
      int bb = m >> 11, s = m & 2047;
#pragma unroll
      for (int nt = 0; nt < 8; ++nt) {
        int n = n0 + nt * 16 + ln, hh = n >> 6, dd = n & 63;
        v8bf pk;
#pragma unroll
        for (int v = 0; v < 8; ++v) pk[v] = (bf16)acc[mi][nt][v];
        *(v8bf*)&o[(((size_t)bb * 16 + hh) * 64 + dd) * 2048 + s] = pk;
      }
    }
  } else {                               // f32 [B*S][1024]
    float* o = (float*)outp;
#pragma unroll
    for (int mi = 0; mi < 2; ++mi)
#pragma unroll
      for (int nt = 0; nt < 8; ++nt) {
        int n = n0 + nt * 16 + ln;
#pragma unroll
        for (int v = 0; v < 8; ++v) {
          int m = m0 + wid * 32 + mi * 16 + v + 8 * half;
          o[(size_t)m * 1024 + n] = acc[mi][nt][v];
        }
      }
  }
}

// ---------------------------------------------------------------------------
// Flash attention, no 1/sqrt(d) scaling (Pix2Struct).
// grid = 512 (32 (b,h) x 16 q-blocks of 128), 256 threads (8 waves x 16 rows).
// Q,K: [b][h][s][64] bf16; Vt: [b][h][64][s] bf16. ctx out: bf16 [B][S][1024].
// ---------------------------------------------------------------------------
__global__ __launch_bounds__(256) void flash_attn(const bf16* __restrict__ Q,
                                                  const bf16* __restrict__ K,
                                                  const bf16* __restrict__ Vt,
                                                  const float* __restrict__ amask,
                                                  bf16* __restrict__ ctx) {
  __shared__ __align__(16) bf16 Qs[128][72];
  __shared__ __align__(16) bf16 Ks[64][72];
  __shared__ __align__(16) bf16 Vs[64][72];   // [dd][key]
  __shared__ __align__(16) bf16 Ps[128][72];  // per-wave 16-row regions
  const int t  = threadIdx.x;
  const int bh = blockIdx.x >> 4;
  const int qb = blockIdx.x & 15;
  const int bb = bh >> 4;
  const int hh = bh & 15;
  const bf16* Qb = Q + (size_t)bh * 2048 * 64 + (size_t)qb * 128 * 64;
  const bf16* Kb = K + (size_t)bh * 2048 * 64;
  const bf16* Vb = Vt + (size_t)bh * 64 * 2048;
  const int wid = t >> 5, lane = t & 31, ln = lane & 15, half = lane >> 4;

  {  // Q tile 128x64 (async)
    int r = t >> 1, c0 = (t & 1) * 32;
    const bf16* g = Qb + r * 64 + c0;
    cp16(&Qs[r][c0],      g);
    cp16(&Qs[r][c0 + 8],  g + 8);
    cp16(&Qs[r][c0 + 16], g + 16);
    cp16(&Qs[r][c0 + 24], g + 24);
  }

  float m_run[8], l_run[8];
#pragma unroll
  for (int v = 0; v < 8; ++v) { m_run[v] = NEGMAX; l_run[v] = 0.f; }
  v8f octx[4] = {};

  for (int kb = 0; kb < 32; ++kb) {
    __syncthreads();
    {  // K tile 64x64 (s-major) and V tile 64x64 (d-major), async
      int r = t >> 2, c0 = (t & 3) * 16;
      const bf16* gk = Kb + (size_t)(kb * 64 + r) * 64 + c0;
      cp16(&Ks[r][c0],     gk);
      cp16(&Ks[r][c0 + 8], gk + 8);
      const bf16* gv = Vb + (size_t)r * 2048 + kb * 64 + c0;
      cp16(&Vs[r][c0],     gv);
      cp16(&Vs[r][c0 + 8], gv + 8);
    }
    wait_async0();
    __syncthreads();

    // S = Q * K^T  (16 queries x 64 keys per wave)
    v8f sacc[4] = {};
#pragma unroll
    for (int kc = 0; kc < 2; ++kc) {
      v16bf a = frag_a(&Qs[0][0], 72, wid * 16 + ln, kc * 32, half);
      v16bf bfr[4];
#pragma unroll
      for (int nt = 0; nt < 4; ++nt)
        bfr[nt] = frag_b(&Ks[0][0], 72, nt * 16 + ln, kc * 32, half);
#pragma unroll
      for (int nt = 0; nt < 4; ++nt) sacc[nt] = wmma_bf16(a, bfr[nt], sacc[nt]);
    }

    // position bias: pb = 1 - mask; where pb==1 -> -FLT_MAX; clamp at -FLT_MAX
#pragma unroll
    for (int nt = 0; nt < 4; ++nt) {
      int keyg = kb * 64 + nt * 16 + ln;
      float mv = amask[bb * 2048 + keyg];
      float pbv = 1.f - mv;
      float bias = (pbv == 1.f) ? NEGMAX : pbv;
#pragma unroll
      for (int v = 0; v < 8; ++v) sacc[nt][v] = fmaxf(sacc[nt][v] + bias, NEGMAX);
    }

    // online softmax: rows striped over 16 lanes of each half-wave
    float mn[8], sc[8], rs[8];
#pragma unroll
    for (int v = 0; v < 8; ++v) {
      float x = fmaxf(fmaxf(sacc[0][v], sacc[1][v]), fmaxf(sacc[2][v], sacc[3][v]));
#pragma unroll
      for (int mk = 1; mk < 16; mk <<= 1) x = fmaxf(x, __shfl_xor(x, mk, 32));
      mn[v] = fmaxf(m_run[v], x);
      sc[v] = __expf(m_run[v] - mn[v]);
      m_run[v] = mn[v];
      rs[v] = 0.f;
    }
#pragma unroll
    for (int j = 0; j < 4; ++j)
#pragma unroll
      for (int v = 0; v < 8; ++v) octx[j][v] *= sc[v];
#pragma unroll
    for (int nt = 0; nt < 4; ++nt)
#pragma unroll
      for (int v = 0; v < 8; ++v) {
        float e = __expf(sacc[nt][v] - mn[v]);
        sacc[nt][v] = e;
        rs[v] += e;
      }
#pragma unroll
    for (int v = 0; v < 8; ++v) {
      float s = rs[v];
#pragma unroll
      for (int mk = 1; mk < 16; mk <<= 1) s += __shfl_xor(s, mk, 32);
      l_run[v] = l_run[v] * sc[v] + s;
    }

    // stage P as bf16 in this wave's LDS rows (C-layout -> A-layout transpose)
#pragma unroll
    for (int nt = 0; nt < 4; ++nt)
#pragma unroll
      for (int v = 0; v < 8; ++v)
        Ps[wid * 16 + v + 8 * half][nt * 16 + ln] = (bf16)sacc[nt][v];
    __syncthreads();

    // ctx += P * V  (V pre-transposed -> B-operand frags are contiguous)
#pragma unroll
    for (int kc = 0; kc < 2; ++kc) {
      v16bf a = frag_a(&Ps[0][0], 72, wid * 16 + ln, kc * 32, half);
      v16bf bfr[4];
#pragma unroll
      for (int j = 0; j < 4; ++j)
        bfr[j] = frag_b(&Vs[0][0], 72, j * 16 + ln, kc * 32, half);
#pragma unroll
      for (int j = 0; j < 4; ++j) octx[j] = wmma_bf16(a, bfr[j], octx[j]);
    }
  }

  // normalize and store ctx as bf16 [B][S][1024] (h = head*64 + dd)
#pragma unroll
  for (int v = 0; v < 8; ++v) {
    float inv = l_run[v] > 0.f ? 1.f / l_run[v] : 0.f;
    int s = qb * 128 + wid * 16 + v + 8 * half;
    size_t base = ((size_t)bb * 2048 + s) * 1024 + hh * 64;
#pragma unroll
    for (int j = 0; j < 4; ++j)
      ctx[base + j * 16 + ln] = (bf16)(octx[j][v] * inv);
  }
}

// position_bias output: 1 - mask, [2,1,1,2048]
__global__ __launch_bounds__(256) void pb_kernel(const float* __restrict__ amask,
                                                 float* __restrict__ pb) {
  int i = blockIdx.x * 256 + threadIdx.x;
  if (i < 4096) pb[i] = 1.f - amask[i];
}

// ---------------------------------------------------------------------------
extern "C" void kernel_launch(void* const* d_in, const int* in_sizes, int n_in,
                              void* d_out, int out_size, void* d_ws, size_t ws_size,
                              hipStream_t stream) {
  (void)in_sizes; (void)n_in; (void)out_size; (void)ws_size;
  const float* hs = (const float*)d_in[0];
  const float* am = (const float*)d_in[1];
  const float* wq = (const float*)d_in[2];
  const float* wk = (const float*)d_in[3];
  const float* wv = (const float*)d_in[4];
  const float* wo = (const float*)d_in[5];
  float* out = (float*)d_out;

  // workspace layout (bf16 elements), ~48 MB total
  bf16* Xb  = (bf16*)d_ws;             // 4096*1024
  bf16* Wqb = Xb  + 4096u * 1024u;     // 1024*1024 each
  bf16* Wkb = Wqb + 1024u * 1024u;
  bf16* Wvb = Wkb + 1024u * 1024u;
  bf16* Wob = Wvb + 1024u * 1024u;
  bf16* Qw  = Wob + 1024u * 1024u;     // [2][16][2048][64]
  bf16* Kw  = Qw  + 4096u * 1024u;
  bf16* Vtw = Kw  + 4096u * 1024u;     // [2][16][64][2048]
  bf16* Cw  = Vtw + 4096u * 1024u;     // [2][2048][1024]

  cvt_f32_bf16<<<4096, 256, 0, stream>>>(hs, Xb);
  cvt_f32_bf16<<<1024, 256, 0, stream>>>(wq, Wqb);
  cvt_f32_bf16<<<1024, 256, 0, stream>>>(wk, Wkb);
  cvt_f32_bf16<<<1024, 256, 0, stream>>>(wv, Wvb);
  cvt_f32_bf16<<<1024, 256, 0, stream>>>(wo, Wob);

  gemm_x_wT<<<128, 256, 0, stream>>>(Xb, Wqb, Qw, 0);
  gemm_x_wT<<<128, 256, 0, stream>>>(Xb, Wkb, Kw, 0);
  gemm_x_wT<<<128, 256, 0, stream>>>(Xb, Wvb, Vtw, 1);

  flash_attn<<<512, 256, 0, stream>>>(Qw, Kw, Vtw, am, Cw);

  gemm_x_wT<<<128, 256, 0, stream>>>(Cw, Wob, out, 2);
  pb_kernel<<<16, 256, 0, stream>>>(am, out + 4096u * 1024u);
}
